// GNNLayer_68453188764092
// MI455X (gfx1250) — compile-verified, matrix operand order checked by reading, not run on hardware
//
#include <hip/hip_runtime.h>
#include <hip/hip_bf16.h>
#include <cstdint>
#include <cstddef>

// Problem constants (match reference)
#define H_    8
#define HID_  256
#define EDIM_ 128
#define NROW_ 20000
#define NTOK_ 400000
#define NE_   400000

typedef __attribute__((ext_vector_type(16))) _Float16 v16h;
typedef __attribute__((ext_vector_type(8)))  float    v8f;

#define SA_STRIDE 40  // halfs per LDS row (80B: 16B-aligned, staggered banks)

// ---------------------------------------------------------------------------
// Elementwise f32 -> f16 conversion
// ---------------------------------------------------------------------------
__global__ void cvt_f32_to_f16(const float* __restrict__ in,
                               _Float16* __restrict__ out, size_t n) {
  size_t i = (size_t)blockIdx.x * blockDim.x + threadIdx.x;
  if (i < n) out[i] = (_Float16)in[i];
}

// W[K,N] row-major (f32) -> Wt[N,K] row-major (f16). Tiny matrices (<=64K elems).
__global__ void transpose_w_f16(const float* __restrict__ W,
                                _Float16* __restrict__ Wt, int K, int N) {
  int t = blockIdx.x * blockDim.x + threadIdx.x;
  if (t < K * N) {
    int k = t / N, n = t % N;
    Wt[(size_t)n * K + k] = (_Float16)W[t];
  }
}

__global__ void init_softmax_state(unsigned* __restrict__ mmax,
                                   float* __restrict__ ssum, int n) {
  int i = blockIdx.x * blockDim.x + threadIdx.x;
  if (i < n) { mmax[i] = 0x007FFFFFu; /* encode(-inf) */ ssum[i] = 0.0f; }
}

// ---------------------------------------------------------------------------
// WMMA GEMM: C[M,N] = A[M,K](f16) * B[K,N](f16, given as Bt[N,K]) + bias[N]
// Block: 256 threads (8 waves), tile 128(M) x 64(N); wave tile 32x32.
// K assumed multiple of 32, N multiple of 64; ragged M guarded.
// ---------------------------------------------------------------------------
__global__ __launch_bounds__(256)
void gemm_wmma_f16(const _Float16* __restrict__ A, const _Float16* __restrict__ Bt,
                   const float* __restrict__ bias, float* __restrict__ Cmat,
                   int M, int N, int K) {
  __shared__ _Float16 sA[128 * SA_STRIDE];
  __shared__ _Float16 sB[64 * SA_STRIDE];
  const int tid  = threadIdx.x;
  const int lane = tid & 31;
  const int wave = tid >> 5;
  const int wm = (wave & 3) * 32;     // wave M-offset inside block tile
  const int wn = (wave >> 2) * 32;    // wave N-offset inside block tile
  const int row0 = blockIdx.x * 128;
  const int col0 = blockIdx.y * 64;

  v8f acc00 = {}, acc01 = {}, acc10 = {}, acc11 = {};

  // 16x32 f16 fragment layout (ISA 7.12.2): lane L -> row (L&15),
  // K-halves: [koff..koff+7] in regs 0..3, [koff+16..koff+23] in regs 4..7.
  const int r    = lane & 15;
  const int koff = (lane >> 4) * 8;

  for (int k0 = 0; k0 < K; k0 += 32) {
    // Stage A tile: 128 rows x 32 halfs, 16B vector chunks, 2 per thread.
    #pragma unroll
    for (int i = 0; i < 2; ++i) {
      int c  = tid + i * 256;
      int ar = c >> 2;
      int ac = (c & 3) * 8;
      int grow = row0 + ar;
      uint4 val = make_uint4(0u, 0u, 0u, 0u);
      if (grow < M) val = *(const uint4*)(A + (size_t)grow * K + k0 + ac);
      *(uint4*)(&sA[ar * SA_STRIDE + ac]) = val;
    }
    // Stage Bt tile: 64 rows x 32 halfs, 1 chunk per thread (N always full).
    {
      int br = tid >> 2;
      int bc = (tid & 3) * 8;
      uint4 val = *(const uint4*)(Bt + (size_t)(col0 + br) * K + k0 + bc);
      *(uint4*)(&sB[br * SA_STRIDE + bc]) = val;
    }
    __syncthreads();

    v16h a0, a1, b0, b1;
    const _Float16* pa0 = &sA[(wm + r)      * SA_STRIDE + koff];
    const _Float16* pa1 = &sA[(wm + 16 + r) * SA_STRIDE + koff];
    const _Float16* pb0 = &sB[(wn + r)      * SA_STRIDE + koff];
    const _Float16* pb1 = &sB[(wn + 16 + r) * SA_STRIDE + koff];
    #pragma unroll
    for (int i = 0; i < 8; ++i) {
      a0[i] = pa0[i]; a0[8 + i] = pa0[16 + i];
      a1[i] = pa1[i]; a1[8 + i] = pa1[16 + i];
      b0[i] = pb0[i]; b0[8 + i] = pb0[16 + i];
      b1[i] = pb1[i]; b1[8 + i] = pb1[16 + i];
    }
    acc00 = __builtin_amdgcn_wmma_f32_16x16x32_f16(false, a0, false, b0, (short)0, acc00, false, false);
    acc01 = __builtin_amdgcn_wmma_f32_16x16x32_f16(false, a0, false, b1, (short)0, acc01, false, false);
    acc10 = __builtin_amdgcn_wmma_f32_16x16x32_f16(false, a1, false, b0, (short)0, acc10, false, false);
    acc11 = __builtin_amdgcn_wmma_f32_16x16x32_f16(false, a1, false, b1, (short)0, acc11, false, false);
    __syncthreads();
  }

  // C/D layout: lanes 0-15 -> rows 0..7 (reg idx), lanes 16-31 -> rows 8..15.
  const int subr = (lane >> 4) * 8;
  const int subc = lane & 15;
  #pragma unroll
  for (int i = 0; i < 2; ++i) {
    #pragma unroll
    for (int j = 0; j < 2; ++j) {
      v8f acc = (i == 0) ? (j == 0 ? acc00 : acc01) : (j == 0 ? acc10 : acc11);
      int col = col0 + wn + j * 16 + subc;
      float bb = bias ? bias[col] : 0.0f;
      #pragma unroll
      for (int vv = 0; vv < 8; ++vv) {
        int rr = row0 + wm + i * 16 + subr + vv;
        if (rr < M) Cmat[(size_t)rr * N + col] = acc[vv] + bb;
      }
    }
  }
}

// ---------------------------------------------------------------------------
// Edge phase. One wave32 per edge; each lane owns 8 contiguous channels,
// 4 lanes per head (C=32).
// ---------------------------------------------------------------------------
__global__ __launch_bounds__(256)
void edge_alpha(const int* __restrict__ src, const int* __restrict__ dst,
                const float* __restrict__ q, const float* __restrict__ k,
                const float* __restrict__ ebuf, float* __restrict__ alpha,
                unsigned* __restrict__ mmax, int nE) {
  int eid = blockIdx.x * 8 + (threadIdx.x >> 5);
  if (eid >= nE) return;
  int lane = threadIdx.x & 31;
  int s = src[eid], d = dst[eid];
  const float4* q4 = (const float4*)(q + (size_t)d * HID_);
  const float4* k4 = (const float4*)(k + (size_t)s * HID_);
  const float4* e4 = (const float4*)(ebuf + (size_t)eid * HID_);
  float4 qa = q4[lane * 2], qb = q4[lane * 2 + 1];
  float4 ka = k4[lane * 2], kb = k4[lane * 2 + 1];
  float4 ea = e4[lane * 2], eb = e4[lane * 2 + 1];
  float acc = qa.x * (ka.x + ea.x) + qa.y * (ka.y + ea.y)
            + qa.z * (ka.z + ea.z) + qa.w * (ka.w + ea.w)
            + qb.x * (kb.x + eb.x) + qb.y * (kb.y + eb.y)
            + qb.z * (kb.z + eb.z) + qb.w * (kb.w + eb.w);
  acc += __shfl_xor(acc, 1, 32);
  acc += __shfl_xor(acc, 2, 32);
  if ((lane & 3) == 0) {
    int h = lane >> 2;
    float al = acc * 0.17677669529663687f;  // 1/sqrt(32)
    alpha[(size_t)eid * H_ + h] = al;
    // monotonic float->uint encoding for atomic segment-max
    unsigned ub  = __float_as_uint(al);
    unsigned enc = (ub & 0x80000000u) ? ~ub : (ub | 0x80000000u);
    atomicMax(mmax + (size_t)d * H_ + h, enc);
  }
}

__global__ void edge_exp(const int* __restrict__ dst, float* __restrict__ alpha,
                         const unsigned* __restrict__ mmax,
                         float* __restrict__ ssum, int nEH) {
  int t = blockIdx.x * blockDim.x + threadIdx.x;
  if (t >= nEH) return;
  int h = t & 7;
  int eid = t >> 3;
  int d = dst[eid];
  unsigned u = mmax[(size_t)d * H_ + h];
  float mv = (u & 0x80000000u) ? __uint_as_float(u & 0x7fffffffu)
                               : __uint_as_float(~u);
  float a = __expf(alpha[t] - mv);
  alpha[t] = a;
  atomicAdd(ssum + (size_t)d * H_ + h, a);
}

__global__ __launch_bounds__(256)
void edge_agg(const int* __restrict__ src, const int* __restrict__ dst,
              const float* __restrict__ v, const float* __restrict__ ebuf,
              const float* __restrict__ alpha, const float* __restrict__ ssum,
              float* __restrict__ out, int nE) {
  int eid = blockIdx.x * 8 + (threadIdx.x >> 5);
  if (eid >= nE) return;
  int lane = threadIdx.x & 31;
  int s = src[eid], d = dst[eid];
  int h = lane >> 2;
  float coef = alpha[(size_t)eid * H_ + h] / (ssum[(size_t)d * H_ + h] + 1e-16f);
  const float4* v4 = (const float4*)(v + (size_t)s * HID_);
  const float4* e4 = (const float4*)(ebuf + (size_t)eid * HID_);
  float4 va = v4[lane * 2], vb = v4[lane * 2 + 1];
  float4 ea = e4[lane * 2], eb = e4[lane * 2 + 1];
  float* op = out + (size_t)d * HID_ + lane * 8;
  atomicAdd(op + 0, (va.x + ea.x) * coef);
  atomicAdd(op + 1, (va.y + ea.y) * coef);
  atomicAdd(op + 2, (va.z + ea.z) * coef);
  atomicAdd(op + 3, (va.w + ea.w) * coef);
  atomicAdd(op + 4, (vb.x + eb.x) * coef);
  atomicAdd(op + 5, (vb.y + eb.y) * coef);
  atomicAdd(op + 6, (vb.z + eb.z) * coef);
  atomicAdd(op + 7, (vb.w + eb.w) * coef);
}

// ---------------------------------------------------------------------------
// Fused residual + LayerNorm: out = LN(xin + agg) * g + b. One wave per row.
// ---------------------------------------------------------------------------
__global__ __launch_bounds__(256)
void ln_fused(const float* __restrict__ xin, const float* __restrict__ agg,
              const float* __restrict__ g, const float* __restrict__ b,
              float* __restrict__ out, int rows) {
  int row = blockIdx.x * 8 + (threadIdx.x >> 5);
  if (row >= rows) return;
  int lane = threadIdx.x & 31;
  const float4* x4 = (const float4*)(xin + (size_t)row * HID_);
  const float4* a4 = (const float4*)(agg + (size_t)row * HID_);
  float4 xa = x4[lane * 2], xb = x4[lane * 2 + 1];
  float4 aa = a4[lane * 2], ab = a4[lane * 2 + 1];
  float vals[8] = { xa.x + aa.x, xa.y + aa.y, xa.z + aa.z, xa.w + aa.w,
                    xb.x + ab.x, xb.y + ab.y, xb.z + ab.z, xb.w + ab.w };
  float sum = 0.0f;
  #pragma unroll
  for (int i = 0; i < 8; ++i) sum += vals[i];
  #pragma unroll
  for (int o = 1; o < 32; o <<= 1) sum += __shfl_xor(sum, o, 32);
  float mu = sum * (1.0f / 256.0f);
  float vs = 0.0f;
  #pragma unroll
  for (int i = 0; i < 8; ++i) { float dd = vals[i] - mu; vs += dd * dd; }
  #pragma unroll
  for (int o = 1; o < 32; o <<= 1) vs += __shfl_xor(vs, o, 32);
  float inv = rsqrtf(vs * (1.0f / 256.0f) + 1e-5f);
  int base = lane * 8;
  float* op = out + (size_t)row * HID_ + base;
  #pragma unroll
  for (int i = 0; i < 8; ++i)
    op[i] = (vals[i] - mu) * inv * g[base + i] + b[base + i];
}

// ---------------------------------------------------------------------------
// Host launch
// ---------------------------------------------------------------------------
extern "C" void kernel_launch(void* const* d_in, const int* in_sizes, int n_in,
                              void* d_out, int out_size, void* d_ws, size_t ws_size,
                              hipStream_t stream) {
  (void)in_sizes; (void)n_in; (void)out_size; (void)ws_size;
  const float* row_x   = (const float*)d_in[0];
  const float* token_x = (const float*)d_in[1];
  const float* ea_t2r  = (const float*)d_in[2];
  const float* ea_r2t  = (const float*)d_in[3];
  const int*   src_t2r = (const int*)d_in[4];
  const int*   dst_t2r = (const int*)d_in[5];
  const int*   src_r2t = (const int*)d_in[6];
  const int*   dst_r2t = (const int*)d_in[7];
  // t2r weights: 8..16 ; r2t weights: 17..25 ; LN params: 26..29
  const float* row_ln_g = (const float*)d_in[26];
  const float* row_ln_b = (const float*)d_in[27];
  const float* tok_ln_g = (const float*)d_in[28];
  const float* tok_ln_b = (const float*)d_in[29];

  float* out_row = (float*)d_out;
  float* out_tok = out_row + (size_t)NROW_ * HID_;

  // ---- workspace carve-out (256B aligned) ----
  char* wp = (char*)d_ws;
  auto carve = [&](size_t bytes) {
    void* r = wp;
    wp += (bytes + 255) & ~(size_t)255;
    return r;
  };
  _Float16* hx_tok = (_Float16*)carve((size_t)NTOK_ * HID_ * 2);
  _Float16* hx_row = (_Float16*)carve((size_t)NROW_ * HID_ * 2);
  _Float16* hattr  = (_Float16*)carve((size_t)NE_ * EDIM_ * 2);
  _Float16* hWq = (_Float16*)carve((size_t)HID_ * HID_ * 2);
  _Float16* hWk = (_Float16*)carve((size_t)HID_ * HID_ * 2);
  _Float16* hWv = (_Float16*)carve((size_t)HID_ * HID_ * 2);
  _Float16* hWs = (_Float16*)carve((size_t)HID_ * HID_ * 2);
  _Float16* hWe = (_Float16*)carve((size_t)HID_ * EDIM_ * 2);
  float*    qbuf   = (float*)carve((size_t)NTOK_ * HID_ * 4);
  float*    kbuf   = (float*)carve((size_t)NTOK_ * HID_ * 4);
  float*    vbuf   = (float*)carve((size_t)NTOK_ * HID_ * 4);
  float*    ebuf   = (float*)carve((size_t)NE_ * HID_ * 4);
  float*    aggbuf = (float*)carve((size_t)NTOK_ * HID_ * 4);
  float*    alphab = (float*)carve((size_t)NE_ * H_ * 4);
  unsigned* mmax   = (unsigned*)carve((size_t)NTOK_ * H_ * 4);
  float*    ssum   = (float*)carve((size_t)NTOK_ * H_ * 4);

  auto cvt = [&](const float* in, _Float16* out, size_t n) {
    int blocks = (int)((n + 255) / 256);
    cvt_f32_to_f16<<<blocks, 256, 0, stream>>>(in, out, n);
  };
  auto gemm = [&](const _Float16* A, const _Float16* Bt, const float* bias,
                  float* C, int M, int N, int K) {
    dim3 grid((M + 127) / 128, N / 64);
    gemm_wmma_f16<<<grid, 256, 0, stream>>>(A, Bt, bias, C, M, N, K);
  };

  auto run_pass = [&](const _Float16* hsrc, const _Float16* hdst,
                      const float* eattr, const int* src, const int* dst,
                      int Nsrc, int Ndst, int widx,
                      const float* xdst_f32, const float* lng, const float* lnb,
                      float* out_region) {
    const float* Wq = (const float*)d_in[widx + 0];
    const float* bq = (const float*)d_in[widx + 1];
    const float* Wk = (const float*)d_in[widx + 2];
    const float* bk = (const float*)d_in[widx + 3];
    const float* Wv = (const float*)d_in[widx + 4];
    const float* bv = (const float*)d_in[widx + 5];
    const float* We = (const float*)d_in[widx + 6];
    const float* Ws = (const float*)d_in[widx + 7];
    const float* bs = (const float*)d_in[widx + 8];

    cvt(eattr, hattr, (size_t)NE_ * EDIM_);
    transpose_w_f16<<<(HID_ * HID_ + 255) / 256, 256, 0, stream>>>(Wq, hWq, HID_, HID_);
    transpose_w_f16<<<(HID_ * HID_ + 255) / 256, 256, 0, stream>>>(Wk, hWk, HID_, HID_);
    transpose_w_f16<<<(HID_ * HID_ + 255) / 256, 256, 0, stream>>>(Wv, hWv, HID_, HID_);
    transpose_w_f16<<<(HID_ * HID_ + 255) / 256, 256, 0, stream>>>(Ws, hWs, HID_, HID_);
    transpose_w_f16<<<(EDIM_ * HID_ + 255) / 256, 256, 0, stream>>>(We, hWe, EDIM_, HID_);

    gemm(hdst,  hWq, bq,      qbuf,   Ndst, HID_, HID_);
    gemm(hsrc,  hWk, bk,      kbuf,   Nsrc, HID_, HID_);
    gemm(hsrc,  hWv, bv,      vbuf,   Nsrc, HID_, HID_);
    gemm(hattr, hWe, nullptr, ebuf,   NE_,  HID_, EDIM_);
    gemm(hdst,  hWs, bs,      aggbuf, Ndst, HID_, HID_);  // seed with skip

    int nstate = Ndst * H_;
    init_softmax_state<<<(nstate + 255) / 256, 256, 0, stream>>>(mmax, ssum, nstate);
    edge_alpha<<<(NE_ + 7) / 8, 256, 0, stream>>>(src, dst, qbuf, kbuf, ebuf, alphab, mmax, NE_);
    edge_exp<<<(NE_ * H_ + 255) / 256, 256, 0, stream>>>(dst, alphab, mmax, ssum, NE_ * H_);
    edge_agg<<<(NE_ + 7) / 8, 256, 0, stream>>>(src, dst, vbuf, ebuf, alphab, ssum, aggbuf, NE_);
    ln_fused<<<(Ndst + 7) / 8, 256, 0, stream>>>(xdst_f32, aggbuf, lng, lnb, out_region, Ndst);
  };

  // node feature conversions
  cvt(token_x, hx_tok, (size_t)NTOK_ * HID_);
  cvt(row_x,   hx_row, (size_t)NROW_ * HID_);

  // Pass 1: token -> row (src=tokens, dst=rows); output = new row_x
  run_pass(hx_tok, hx_row, ea_t2r, src_t2r, dst_t2r, NTOK_, NROW_, 8,
           row_x, row_ln_g, row_ln_b, out_row);

  // Pass 2 uses the updated row features as source
  cvt(out_row, hx_row, (size_t)NROW_ * HID_);
  run_pass(hx_row, hx_tok, ea_r2t, src_r2t, dst_r2t, NROW_, NTOK_, 17,
           token_x, tok_ln_g, tok_ln_b, out_tok);
}
